// VectorQuantizer_40862318854532
// MI455X (gfx1250) — matrix operand order repaired and checked by reading, not compiled, not run on hardware
//
#include <hip/hip_runtime.h>

// CDNA5 / gfx1250 vector-quantizer:
//   argmin_k ||x - w_k||^2  ==  argmin_k ( w2[k] - 2 * x . w[k] )
//   cross term: v_wmma_f32_16x16x32_bf16; codebook tiles staged in LDS via
//   global_load_async_to_lds_b128 (double-buffered, ASYNCcnt); w2 + gather fp32.

typedef __attribute__((ext_vector_type(16))) __bf16 v16bf;
typedef __attribute__((ext_vector_type(8)))  float  v8f;

__device__ __forceinline__ unsigned short f2bf(float f) {
  // round-to-nearest-even fp32 -> bf16
  unsigned int u = __float_as_uint(f);
  u += 0x7FFFu + ((u >> 16) & 1u);
  return (unsigned short)(u >> 16);
}

// gfx1250 async global->LDS copy, tracked by ASYNCcnt (ISA 15.18.3 op 98)
__device__ __forceinline__ void async_ld_b128(unsigned lds_off, const void* g) {
  asm volatile("global_load_async_to_lds_b128 %0, %1, off"
               :: "v"(lds_off), "v"(g) : "memory");
}
__device__ __forceinline__ void wait_async0() {
  asm volatile("s_wait_asynccnt 0" ::: "memory");
}

// ---------------------------------------------------------------------------
// Pass 1: codebook fp32 [1024,256] -> bf16 [1024,256] + exact fp32 w2[1024]
// ---------------------------------------------------------------------------
__global__ void __launch_bounds__(256)
vq_prep(const float* __restrict__ w, unsigned short* __restrict__ wbf,
        float* __restrict__ w2) {
  const int row = blockIdx.x;      // 1024 rows
  const int t   = threadIdx.x;     // 256 threads = D
  const float v = w[(size_t)row * 256 + t];
  wbf[(size_t)row * 256 + t] = f2bf(v);

  __shared__ float red[256];
  red[t] = v * v;
  __syncthreads();
  #pragma unroll
  for (int s = 128; s > 0; s >>= 1) {
    if (t < s) red[t] += red[t + s];
    __syncthreads();
  }
  if (t == 0) w2[row] = red[0];
}

// ---------------------------------------------------------------------------
// Pass 2: per-wave 16-row tile; block-shared double-buffered LDS staging of
//         codebook tiles; bf16 WMMA scan; running argmin; fp32 gather.
// ---------------------------------------------------------------------------
__global__ void __launch_bounds__(256)
vq_argmin_gather(const float* __restrict__ x,
                 const unsigned short* __restrict__ wbf,
                 const float* __restrict__ w2,
                 const float* __restrict__ wfp,
                 float* __restrict__ out) {
  constexpr int D = 256, K = 1024, HW = 1024;     // H*W = 32*32
  constexpr int RSTR = 264;                       // LDS row stride (ushort): 528B,
                                                  // +16B pad -> bank-conflict-free halves
  __shared__ __align__(16) unsigned short bstage[2][16 * RSTR];  // 2 x 8.25 KB
  __shared__ int sidx[8][16];

  const int tid  = threadIdx.x;
  const int lane = tid & 31;
  const int wave = tid >> 5;
  const int col  = lane & 15;                  // row-in-tile (A) / column (B,C)
  const int hi   = lane >> 4;                  // half-wave select
  const int tile = blockIdx.x * 8 + wave;      // 16-row tile id, 2048 tiles
  const int row0 = tile * 16;

  // ---- A fragments: 16 rows x 256 d of x, bf16, kept in registers --------
  // A (16x32), k order chosen here: lane holds row m = lane&15; lanes 0-15 cover
  // K = 0..15, lanes 16-31 cover K = 16..31 (2 per VGPR), matching B.
  const int  myrow = row0 + col;
  const int  ab    = myrow >> 10;              // batch
  const int  ahw   = myrow & 1023;             // h*W + w
  const float* xrow = x + ((size_t)ab * D) * HW + ahw;   // elem d: xrow[d*HW]
  const int  khalf = hi * 16;

  v16bf a[8];
  #pragma unroll
  for (int c = 0; c < 8; ++c) {
    union { v16bf v; unsigned short s[16]; } u;
    const int base = 32 * c + khalf;
    #pragma unroll
    for (int j = 0; j < 8; ++j)
      u.s[j]     = f2bf(xrow[(size_t)(base + j) * HW]);
    #pragma unroll
    for (int j = 0; j < 8; ++j)
      u.s[8 + j] = f2bf(xrow[(size_t)(base + 8 + j) * HW]);
    a[c] = u.v;
  }

  // ---- async staging: thread tid copies 2 x 16B chunks of a tile ---------
  auto issue_tile = [&](int kt, int buf) {
    const unsigned lbase = (unsigned)(size_t)(&bstage[buf][0]);
    #pragma unroll
    for (int i = 0; i < 2; ++i) {
      const int chunk = tid + i * 256;         // 0..511 (16 rows x 512B)
      const int row   = chunk >> 5;            // code row in tile, 0..15
      const int boff  = (chunk & 31) * 16;     // byte offset within row
      const char* g = (const char*)(wbf + (size_t)(kt * 16 + row) * D) + boff;
      async_ld_b128(lbase + (unsigned)(row * (RSTR * 2) + boff), g);
    }
  };

  float best[8];
  int   bidx[8];
  #pragma unroll
  for (int v = 0; v < 8; ++v) { best[v] = 3.4e38f; bidx[v] = 0; }

  issue_tile(0, 0);

  // B (32x16 bf16), same k order as A: lane holds column n = lane&15; lanes 0-15 cover
  // K=0..15, lanes 16-31 cover K=16..31 -> one 32B LDS read per lane/chunk.
  const int kdbase = hi * 16;
  for (int kt = 0; kt < K / 16; ++kt) {
    wait_async0();        // my chunks of cur tile have landed in LDS
    __syncthreads();      // everyone's chunks landed; prev reads of nxt done
    if (kt + 1 < K / 16) issue_tile(kt + 1, (kt + 1) & 1);

    const unsigned short* blane = &bstage[kt & 1][0] + col * RSTR + kdbase;

    v8f acc = {};
    #pragma unroll
    for (int c = 0; c < 8; ++c) {
      v16bf b = *(const v16bf*)(blane + 32 * c);   // ds_load_b128 x2
      acc = __builtin_amdgcn_wmma_f32_16x16x32_bf16(
          /*neg_a=*/false, a[c], /*neg_b=*/false, b,
          /*c_mod=*/(short)0, acc, /*reuse_a=*/false, /*reuse_b=*/false);
    }

    // C/D layout: lane holds n = lane&15; VGPR v holds row m = v + hi*8.
    const int   code = kt * 16 + col;
    const float w2k  = w2[code];
    #pragma unroll
    for (int v = 0; v < 8; ++v) {
      const float s = w2k - 2.0f * acc[v];
      if (s < best[v]) { best[v] = s; bidx[v] = code; }
    }
  }

  // min-reduce the 16 columns per row across each 16-lane half
  #pragma unroll
  for (int mask = 1; mask <= 8; mask <<= 1) {
    #pragma unroll
    for (int v = 0; v < 8; ++v) {
      const float os = __shfl_xor(best[v], mask, 32);
      const int   oi = __shfl_xor(bidx[v], mask, 32);
      if (os < best[v] || (os == best[v] && oi < bidx[v])) {
        best[v] = os; bidx[v] = oi;
      }
    }
  }

  if (col == 0) {
    #pragma unroll
    for (int v = 0; v < 8; ++v) sidx[wave][hi * 8 + v] = bidx[v];
  }
  __syncthreads();

  // ---- gather: out[b, :, h, w] = wfp[code, :] (exact fp32) ---------------
  const int  code = sidx[wave][col];
  const int  n    = row0 + col;
  const int  ob   = n >> 10;
  const int  ohw  = n & 1023;
  const float* src = wfp + (size_t)code * D;
  float*       dst = out + ((size_t)ob * D) * HW + ohw;
  for (int d0 = 0; d0 < D; d0 += 2) {
    const int d = d0 + hi;               // two d's per iter, 16 lanes each
    dst[(size_t)d * HW] = src[d];        // 64B-coalesced per half-wave
  }
}

// ---------------------------------------------------------------------------
extern "C" void kernel_launch(void* const* d_in, const int* in_sizes, int n_in,
                              void* d_out, int out_size, void* d_ws, size_t ws_size,
                              hipStream_t stream) {
  (void)in_sizes; (void)n_in; (void)out_size; (void)ws_size;
  const float* x = (const float*)d_in[0];   // [32,256,32,32]
  const float* w = (const float*)d_in[1];   // [1024,256]
  float* out = (float*)d_out;               // [32,256,32,32]

  unsigned short* wbf = (unsigned short*)d_ws;                    // 512 KB
  float* w2 = (float*)((char*)d_ws + (size_t)1024 * 256 * 2);     // 4 KB

  vq_prep<<<dim3(1024), dim3(256), 0, stream>>>(w, wbf, w2);
  vq_argmin_gather<<<dim3(256), dim3(256), 0, stream>>>(x, wbf, w2, w, out);
}
